// CausalSelfAttention_83236466196861
// MI455X (gfx1250) — matrix operand order repaired and checked by reading, not compiled, not run on hardware
//
#include <hip/hip_runtime.h>
#include <hip/hip_bf16.h>

typedef __attribute__((ext_vector_type(16))) _Float16 v16h;
typedef __attribute__((ext_vector_type(8)))  _Float16 v8h;
typedef __attribute__((ext_vector_type(4)))  _Float16 v4h;
typedef __attribute__((ext_vector_type(8)))  float    v8f;

#define BATCH 4
#define SEQ   2048
#define EMB   1024
#define NH    16
#define HD    64
#define NQKV  3072

__device__ __forceinline__ v16h combine16(v8h lo, v8h hi){
  v16h r;
#pragma unroll
  for (int e = 0; e < 8; ++e){ r[e] = lo[e]; r[e+8] = hi[e]; }
  return r;
}

// A-matrix fragment (16 x K=32, f16). Lanes 0-15: k={0..7,16..23},
// lanes 16-31: k={8..15,24..31}. `row` points at k=0 of this lane's M-row.
__device__ __forceinline__ v16h load_frag_a(const _Float16* row, int hf){
  v8h lo = *(const v8h*)(row + hf*8);
  v8h hi = *(const v8h*)(row + hf*8 + 16);
  return combine16(lo, hi);
}

// B-matrix fragment (K=32 x 16, f16). Per lane one column (N=lane&15);
// lanes 0-15: k={0..7,16..23}, lanes 16-31: k={8..15,24..31}.
__device__ __forceinline__ v16h load_frag_b(const _Float16* col, int hf){
  v8h lo = *(const v8h*)(col + hf*8);
  v8h hi = *(const v8h*)(col + hf*8 + 16);
  return combine16(lo, hi);
}

__device__ __forceinline__ v8f wmma_f16f32(v16h a, v16h b, v8f c){
  return __builtin_amdgcn_wmma_f32_16x16x32_f16(
      /*neg_a=*/false, a, /*neg_b=*/false, b,
      /*c_mod=*/(short)0, c, /*reuse_a=*/false, /*reuse_b=*/false);
}

// Async global -> LDS 16B copy (ASYNCcnt tracked, bypasses VGPRs).
// lds_byte_off: low 32 bits of a generic pointer to __shared__ == LDS address.
__device__ __forceinline__ void async_cp16(unsigned lds_byte_off, const void* gaddr){
  asm volatile("global_load_async_to_lds_b128 %0, %1, off"
               :: "v"(lds_byte_off), "v"(gaddr)
               : "memory");
}

// ---------------------------------------------------------------------------
// One-time conversion kernels (negligible vs ~100 GFLOP of matrix work):
// make all GEMM operands f16 and pre-transposed so the main loops can use
// pure async byte copies into LDS.
// ---------------------------------------------------------------------------
__global__ __launch_bounds__(256)
void cvt_f32_f16(const float* __restrict__ src, _Float16* __restrict__ dst, int n4){
  int i = blockIdx.x*256 + threadIdx.x;            // chunk of 4 elements
  if (i < n4){
    float4 f = ((const float4*)src)[i];
    v4h h;
    h[0] = (_Float16)f.x; h[1] = (_Float16)f.y;
    h[2] = (_Float16)f.z; h[3] = (_Float16)f.w;
    ((v4h*)dst)[i] = h;
  }
}

// WT[n][k] = (f16) W[k][n];  W is K x N row-major. 32x32 LDS-tiled.
__global__ __launch_bounds__(256)
void transpose_f32_f16(const float* __restrict__ W, _Float16* __restrict__ WT,
                       int K, int N){
  __shared__ float tile[32][33];
  const int n0 = blockIdx.x * 32;
  const int k0 = blockIdx.y * 32;
  const int tx = threadIdx.x & 31;
  const int ty = threadIdx.x >> 5;                 // 0..7
#pragma unroll
  for (int i = 0; i < 4; ++i){
    int k = ty + i*8;
    tile[k][tx] = W[(size_t)(k0 + k)*N + n0 + tx];
  }
  __syncthreads();
#pragma unroll
  for (int i = 0; i < 4; ++i){
    int n = ty + i*8;
    WT[(size_t)(n0 + n)*K + k0 + tx] = (_Float16)tile[tx][n];
  }
}

// ---------------------------------------------------------------------------
// Tiled WMMA GEMM:  C(8192 x N) = A(8192 x 1024) * WT^T, A and WT both f16.
//   tile 128x64, 8 waves (4x2), wave = 32x32 (2x2 WMMA), BK=32.
//   Double-buffered LDS filled via global_load_async_to_lds_b128.
//   QKV_EPI=true:  scatter q[b,h,t,d], k[b,h,t,d], vT[b,h,d,t] as f16.
//   QKV_EPI=false: fp32 row-major store to `out`.
// ---------------------------------------------------------------------------
template<bool QKV_EPI>
__global__ __launch_bounds__(256)
void gemm_wmma(const _Float16* __restrict__ A, const _Float16* __restrict__ WT,
               _Float16* __restrict__ qbuf, _Float16* __restrict__ kbuf,
               _Float16* __restrict__ vtbuf, float* __restrict__ out)
{
  __shared__ alignas(16) _Float16 As[2][128*40];   // stride 40 halves (80B)
  __shared__ alignas(16) _Float16 Bs[2][64*40];    // WT rows: Bs[n][k]

  const int tid   = threadIdx.x;
  const int wid   = tid >> 5;
  const int lane  = tid & 31;
  const int hf    = lane >> 4;
  const int lrow  = lane & 15;
  const int m0    = (wid >> 1) * 32;
  const int n0    = (wid & 1) * 32;
  const int mbase = blockIdx.y * 128;
  const int nbase = blockIdx.x * 64;

  v8f acc[2][2];
#pragma unroll
  for (int i = 0; i < 2; ++i)
#pragma unroll
    for (int j = 0; j < 2; ++j)
#pragma unroll
      for (int r = 0; r < 8; ++r) acc[i][j][r] = 0.f;

  // 3 async instructions per wave per tile: 2 for A (512 chunks), 1 for B (256).
  auto stage = [&](int k0, int b){
#pragma unroll
    for (int i = 0; i < 2; ++i){
      int c8 = i*256 + tid;
      int r  = c8 >> 2;
      int cc = (c8 & 3) * 8;
      async_cp16((unsigned)(uintptr_t)&As[b][r*40 + cc],
                 A + (size_t)(mbase + r)*EMB + k0 + cc);
    }
    {
      int r  = tid >> 2;
      int cc = (tid & 3) * 8;
      async_cp16((unsigned)(uintptr_t)&Bs[b][r*40 + cc],
                 WT + (size_t)(nbase + r)*EMB + k0 + cc);
    }
  };

  stage(0, 0);
  int buf = 0;
  for (int k0 = 0; k0 < EMB; k0 += 32){
    if (k0 + 32 < EMB){
      stage(k0 + 32, buf ^ 1);                       // prefetch next tile
      asm volatile("s_wait_asynccnt 0x3" ::: "memory");  // current tile done
    } else {
      asm volatile("s_wait_asynccnt 0x0" ::: "memory");
    }
    __syncthreads();

    v16h af[2], bf[2];
#pragma unroll
    for (int i = 0; i < 2; ++i) af[i] = load_frag_a(&As[buf][(m0 + i*16 + lrow)*40], hf);
#pragma unroll
    for (int j = 0; j < 2; ++j) bf[j] = load_frag_b(&Bs[buf][(n0 + j*16 + lrow)*40], hf);
#pragma unroll
    for (int i = 0; i < 2; ++i)
#pragma unroll
      for (int j = 0; j < 2; ++j)
        acc[i][j] = wmma_f16f32(af[i], bf[j], acc[i][j]);

    __syncthreads();                                 // all waves done with buf
    buf ^= 1;
  }

  // ---- epilogue ----
#pragma unroll
  for (int i = 0; i < 2; ++i){
#pragma unroll
    for (int j = 0; j < 2; ++j){
      const int gn = nbase + n0 + j*16 + lrow;
      if (QKV_EPI){
        const int which = gn >> 10;                  // 0=q 1=k 2=v
        const int head  = (gn & 1023) >> 6;
        const int d     = gn & 63;
#pragma unroll
        for (int r = 0; r < 8; ++r){
          const int gm = mbase + m0 + i*16 + hf*8 + r;   // C-layout: M = r + 8*hf
          const int b  = gm >> 11;
          const int t  = gm & 2047;
          const int bh = b*NH + head;
          const _Float16 v = (_Float16)acc[i][j][r];
          if      (which == 0) qbuf [((size_t)bh*SEQ + t)*HD + d]  = v;
          else if (which == 1) kbuf [((size_t)bh*SEQ + t)*HD + d]  = v;
          else                 vtbuf[((size_t)bh*HD  + d)*SEQ + t] = v;  // V^T
        }
      } else {
#pragma unroll
        for (int r = 0; r < 8; ++r){
          const int gm = mbase + m0 + i*16 + hf*8 + r;
          out[(size_t)gm*EMB + gn] = acc[i][j][r];
        }
      }
    }
  }
}

// ---------------------------------------------------------------------------
// Flash attention: grid = (B*H, T/64), 128 threads = 4 waves.
// Each wave owns 16 query rows; loops 32-key blocks with online softmax.
// Per block: 4 WMMA (Q*K^T) + 4 WMMA (P*V). P relayout C->A via per-wave LDS.
// ---------------------------------------------------------------------------
__global__ __launch_bounds__(128)
void attn_wmma(const _Float16* __restrict__ qb, const _Float16* __restrict__ kb,
               const _Float16* __restrict__ vt, _Float16* __restrict__ ob)
{
  __shared__ alignas(16) _Float16 Pt[4][16*40];     // per-wave 16x32 P tile

  const int tid   = threadIdx.x;
  const int wid   = tid >> 5;
  const int lane  = tid & 31;
  const int hf    = lane >> 4;
  const int lrow  = lane & 15;
  const int bh    = blockIdx.x;                     // b*16 + head
  const int qbase = blockIdx.y*64 + wid*16;

  // Q fragments, pre-scaled by 1/sqrt(64) = 0.125 (exact in f16)
  const _Float16* qrow = qb + ((size_t)bh*SEQ + qbase + lrow)*HD;
  v16h aq[2];
#pragma unroll
  for (int db = 0; db < 2; ++db){
    v16h f = load_frag_a(qrow + db*32, hf);
#pragma unroll
    for (int e = 0; e < 16; ++e) f[e] = f[e] * (_Float16)0.125f;
    aq[db] = f;
  }

  float mrun[8], lrun[8];
  v8f oacc[4];
#pragma unroll
  for (int r = 0; r < 8; ++r){ mrun[r] = -3.0e38f; lrun[r] = 0.f; }
#pragma unroll
  for (int t = 0; t < 4; ++t)
#pragma unroll
    for (int r = 0; r < 8; ++r) oacc[t][r] = 0.f;

  const _Float16* kbp = kb + (size_t)bh*SEQ*HD;     // [t][d]
  const _Float16* vbp = vt + (size_t)bh*HD*SEQ;     // [d][t]
  const int nblocks = (qbase + 47) >> 5;

  for (int jb = 0; jb < nblocks; ++jb){
    const int key0 = jb*32;

    v8f s[2];
#pragma unroll
    for (int si = 0; si < 2; ++si){
      v8f c;
#pragma unroll
      for (int r = 0; r < 8; ++r) c[r] = 0.f;
#pragma unroll
      for (int db = 0; db < 2; ++db){
        v16h bk = load_frag_b(kbp + (size_t)(key0 + si*16 + lrow)*HD + db*32, hf);
        c = wmma_f16f32(aq[db], bk, c);
      }
      s[si] = c;
    }

    const bool masked = (key0 + 31 > qbase);
    float p0[8], p1[8];
#pragma unroll
    for (int r = 0; r < 8; ++r){
      float v0 = s[0][r], v1 = s[1][r];
      if (masked){
        const int row = qbase + hf*8 + r;
        if (key0 + lrow      > row) v0 = -3.0e38f;
        if (key0 + 16 + lrow > row) v1 = -3.0e38f;
      }
      float mx = fmaxf(v0, v1);
#pragma unroll
      for (int off = 1; off < 16; off <<= 1) mx = fmaxf(mx, __shfl_xor(mx, off, 32));
      const float mn    = fmaxf(mrun[r], mx);
      const float alpha = __expf(mrun[r] - mn);
      const float e0 = __expf(v0 - mn);
      const float e1 = __expf(v1 - mn);
      float rs = e0 + e1;
#pragma unroll
      for (int off = 1; off < 16; off <<= 1) rs += __shfl_xor(rs, off, 32);
      lrun[r] = lrun[r]*alpha + rs;
      mrun[r] = mn;
      p0[r] = e0; p1[r] = e1;
#pragma unroll
      for (int t = 0; t < 4; ++t) oacc[t][r] *= alpha;
    }

    // P: C-layout -> A-layout via per-wave LDS (wave LDS ops are in-order)
    _Float16* pw = Pt[wid];
#pragma unroll
    for (int r = 0; r < 8; ++r){
      pw[(hf*8 + r)*40 + lrow]      = (_Float16)p0[r];
      pw[(hf*8 + r)*40 + 16 + lrow] = (_Float16)p1[r];
    }
    v16h ap = load_frag_a(&pw[lrow*40], hf);

#pragma unroll
    for (int dt = 0; dt < 4; ++dt){
      v16h bv = load_frag_b(vbp + (size_t)(dt*16 + lrow)*SEQ + key0, hf);
      oacc[dt] = wmma_f16f32(ap, bv, oacc[dt]);
    }
  }

  const int b = bh >> 4, h = bh & 15;
#pragma unroll
  for (int dt = 0; dt < 4; ++dt){
    const int d = dt*16 + lrow;
#pragma unroll
    for (int r = 0; r < 8; ++r){
      const int t = qbase + hf*8 + r;
      const float val = oacc[dt][r] / lrun[r];
      ob[(((size_t)b*SEQ + t)*NH + h)*HD + d] = (_Float16)val;
    }
  }
}

// ---------------------------------------------------------------------------
extern "C" void kernel_launch(void* const* d_in, const int* in_sizes, int n_in,
                              void* d_out, int out_size, void* d_ws, size_t ws_size,
                              hipStream_t stream)
{
  const float* x  = (const float*)d_in[0];   // [4,2048,1024]
  const float* Wa = (const float*)d_in[1];   // [1024,3072]
  const float* Wp = (const float*)d_in[2];   // [1024,1024]
  float* out = (float*)d_out;                // [4,2048,1024]

  const size_t NTOK = (size_t)BATCH*SEQ;               // 8192
  const size_t XE   = NTOK*EMB;                        // 8,388,608
  _Float16* xh   = (_Float16*)d_ws;                    // 16 MB
  _Float16* WaT  = xh   + XE;                          // [3072][1024] f16, 6 MB
  _Float16* WpT  = WaT  + (size_t)EMB*NQKV;            // [1024][1024] f16, 2 MB
  _Float16* qbuf = WpT  + (size_t)EMB*EMB;
  _Float16* kbuf = qbuf + XE;
  _Float16* vtb  = kbuf + XE;
  _Float16* obuf = vtb  + XE;                          // total ws ~88 MB

  cvt_f32_f16<<<(int)(XE/4/256), 256, 0, stream>>>(x, xh, (int)(XE/4));
  transpose_f32_f16<<<dim3(NQKV/32, EMB/32), 256, 0, stream>>>(Wa, WaT, EMB, NQKV);
  transpose_f32_f16<<<dim3(EMB/32,  EMB/32), 256, 0, stream>>>(Wp, WpT, EMB, EMB);

  dim3 g1(NQKV/64, (int)(NTOK/128));                   // 48 x 64
  gemm_wmma<true><<<g1, 256, 0, stream>>>(xh, WaT, qbuf, kbuf, vtb, nullptr);

  dim3 g2(BATCH*NH, SEQ/64);                           // 64 x 32
  attn_wmma<<<g2, 128, 0, stream>>>(qbuf, kbuf, vtb, obuf);

  dim3 g3(EMB/64, (int)(NTOK/128));                    // 16 x 64
  gemm_wmma<false><<<g3, 256, 0, stream>>>(obuf, WpT, nullptr, nullptr, nullptr, out);
}
